// CvxpyProjectionLayer_20650202759697
// MI455X (gfx1250) — matrix operand order repaired and checked by reading, not compiled, not run on hardware
//
#include <hip/hip_runtime.h>

// ---------------------------------------------------------------------------
// Batched FISTA QP projection for MI455X (gfx1250).
// One workgroup (6 wave32) per batch element; A (85x80) + A^T resident in LDS
// as f16; all GEMVs via v_wmma_f32_16x16x32_f16 with fp32 accumulation.
// ---------------------------------------------------------------------------

typedef _Float16 half8_t  __attribute__((ext_vector_type(8)));
typedef _Float16 half16_t __attribute__((ext_vector_type(16)));
typedef float    f32x8_t  __attribute__((ext_vector_type(8)));

#define M_DIM       85      // constraint rows
#define N_DIM       80      // variable dim
#define PAD         96      // padded vector length / row count (6 and 5 tiles)
#define PITCH       104     // halves per LDS matrix row: 208B stride -> no bank conflicts
#define NTHREADS    192     // 6 wave32
#define NWAVES      6
#define POWER_IT    30
#define FISTA_IT    200

// A-operand of v_wmma_f32_16x16x32_f16 (16x32 f16 tile, ISA 7.12.2 layout):
// lane group g = lane>>4 ; row M = lane&15 ;
// elements 0..7  = K[k0 + 8g .. +7], elements 8..15 = K[k0+16+8g .. +7].
__device__ __forceinline__ half16_t ld_amat(const _Float16* Mx, int t16, int k0, int lane) {
  const int g  = lane >> 4;
  const int mr = lane & 15;
  const _Float16* base = Mx + (t16 + mr) * PITCH + k0 + g * 8;
  half8_t lo = *(const half8_t*)(base);        // ds_load_b128
  half8_t hi = *(const half8_t*)(base + 16);   // ds_load_b128
  return __builtin_shufflevector(lo, hi, 0,1,2,3,4,5,6,7,8,9,10,11,12,13,14,15);
}

// B-operand (32x16 f16): VGPR j<4 holds packed K{8g+2j,+1}, VGPR j>=4 K{16+8g+2(j-4),+1}
// N = lane-in-group. We replicate the vector across all N columns,
// so each lane-group broadcasts K[k0+8g..+7] and K[k0+16+8g..+7] (same order as A).
__device__ __forceinline__ half16_t ld_bvec(const _Float16* vh, int k0, int lane) {
  const int g = lane >> 4;
  const _Float16* base = vh + k0 + g * 8;
  half8_t lo = *(const half8_t*)(base);        // broadcast ds_load_b128
  half8_t hi = *(const half8_t*)(base + 16);
  return __builtin_shufflevector(lo, hi, 0,1,2,3,4,5,6,7,8,9,10,11,12,13,14,15);
}

// One 16-row GEMV tile: D(16x16) = Mx[t16:t16+16, 0:96] * repl(vec), col 0 = result.
// C/D layout: VGPR r, lanes 0-15 -> row t16+r ; lanes 16-31 -> row t16+8+r.
__device__ __forceinline__ f32x8_t gemv_tile(const _Float16* Mx, const _Float16* vh,
                                             int t16, int lane) {
  f32x8_t acc = {};
  #pragma unroll
  for (int k0 = 0; k0 < PAD; k0 += 32) {
    half16_t a = ld_amat(Mx, t16, k0, lane);
    half16_t b = ld_bvec(vh, k0, lane);
    acc = __builtin_amdgcn_wmma_f32_16x16x32_f16(false, a, false, b,
                                                 (short)0, acc, false, false);
  }
  return acc;
}

__global__ __launch_bounds__(NTHREADS)
void fista_qp_kernel(const float* __restrict__ x_raw, const float* __restrict__ Ag,
                     const float* __restrict__ bg,    const float* __restrict__ lowg,
                     const float* __restrict__ upg,   float* __restrict__ out)
{
  __shared__ alignas(32) _Float16 A16 [PAD * PITCH];   // A   (f16, zero padded)
  __shared__ alignas(32) _Float16 At16[PAD * PITCH];   // A^T (f16, zero padded)
  __shared__ alignas(32) _Float16 vecH[PAD];           // f16 GEMV operand (y or v)
  __shared__ alignas(32) _Float16 rH  [PAD];           // f16 operand for A^T pass
  __shared__ alignas(16) float ys[PAD], xs[PAD], bs[PAD], xr[PAD], los[PAD], ups[PAD];
  __shared__ float scal;                               // inv-norm / step broadcast

  const int tid  = threadIdx.x;
  const int lane = tid & 31;
  const int wave = tid >> 5;
  const int blk  = blockIdx.x;

  // ---- phase 0: zero-fill padded matrices, stage vectors ------------------
  {
    unsigned int* a32  = (unsigned int*)A16;
    unsigned int* at32 = (unsigned int*)At16;
    for (int i = tid; i < (PAD * PITCH) / 2; i += NTHREADS) { a32[i] = 0u; at32[i] = 0u; }
  }
  const float* xrg = x_raw + (size_t)blk * N_DIM;
  const float* bgp = bg    + (size_t)blk * M_DIM;
  const float* lgp = lowg  + (size_t)blk * N_DIM;
  const float* ugp = upg   + (size_t)blk * N_DIM;
  for (int j = tid; j < PAD; j += NTHREADS) {
    bs[j]   = (j < M_DIM) ? bgp[j] : 0.f;
    float v = (j < N_DIM) ? xrg[j] : 0.f;
    xr[j]   = v;
    los[j]  = (j < N_DIM) ? lgp[j] : 0.f;
    ups[j]  = (j < N_DIM) ? ugp[j] : 0.f;
    vecH[j] = (j < N_DIM) ? (_Float16)0.11180339887f : (_Float16)0.f;  // 1/sqrt(80)
    rH[j]   = (_Float16)0.f;
  }
  __syncthreads();

  // ---- phase 1: A (fp32, HBM, read once) -> LDS f16 A and A^T -------------
  const float* Agp = Ag + (size_t)blk * (M_DIM * N_DIM);
  for (int l = tid; l < M_DIM * N_DIM; l += NTHREADS) {
    int i = l / N_DIM;
    int j = l - i * N_DIM;
    _Float16 h = (_Float16)Agp[l];
    A16 [i * PITCH + j] = h;
    At16[j * PITCH + i] = h;
  }
  __syncthreads();

  // ---- phase 2: power iteration for sigma_max(A)^2 ------------------------
  for (int it = 0; it < POWER_IT; ++it) {
    { // w = A v  (6 tiles, one per wave) -> rH (f16)
      f32x8_t acc = gemv_tile(A16, vecH, wave * 16, lane);
      if (lane == 0 || lane == 16) {
        int rb = wave * 16 + (lane ? 8 : 0);
        #pragma unroll
        for (int r = 0; r < 8; ++r) rH[rb + r] = (_Float16)acc[r];  // pad rows give 0
      }
    }
    __syncthreads();
    if (wave < 5) { // v' = A^T w (5 tiles) -> ys (f32, unnormalized)
      f32x8_t acc = gemv_tile(At16, rH, wave * 16, lane);
      if (lane == 0 || lane == 16) {
        int jb = wave * 16 + (lane ? 8 : 0);
        #pragma unroll
        for (int r = 0; r < 8; ++r) ys[jb + r] = acc[r];
      }
    }
    __syncthreads();
    if (tid == 0) {
      float s = 0.f;
      for (int j = 0; j < N_DIM; ++j) s += ys[j] * ys[j];
      scal = 1.f / (sqrtf(s) + 1e-12f);
    }
    __syncthreads();
    float inv = scal;
    for (int j = tid; j < PAD; j += NTHREADS)
      vecH[j] = (j < N_DIM) ? (_Float16)(ys[j] * inv) : (_Float16)0.f;
    __syncthreads();
  }
  { // sigma^2 = ||A v||^2 ; step = 1 / (1 + 2*p*sigma^2), p = 1
    f32x8_t acc = gemv_tile(A16, vecH, wave * 16, lane);
    if (lane == 0 || lane == 16) {
      int rb = wave * 16 + (lane ? 8 : 0);
      #pragma unroll
      for (int r = 0; r < 8; ++r) ys[rb + r] = acc[r];
    }
  }
  __syncthreads();
  if (tid == 0) {
    float s = 0.f;
    for (int i = 0; i < PAD; ++i) s += ys[i] * ys[i];   // rows >= 85 are exactly 0
    scal = 1.f / (1.f + 2.f * s);
  }
  __syncthreads();
  const float step = scal;

  // ---- phase 3: FISTA init -------------------------------------------------
  for (int j = tid; j < PAD; j += NTHREADS) {
    float x0 = 0.f;
    if (j < N_DIM) x0 = fminf(fmaxf(xr[j], los[j]), ups[j]);
    xs[j] = x0;
    ys[j] = x0;
    vecH[j] = (_Float16)x0;
  }
  __syncthreads();

  // ---- phase 4: 200 FISTA iterations, 2 barriers each ---------------------
  float t = 1.f;
  for (int it = 0; it < FISTA_IT; ++it) {
    { // r = relu(A y - b), fused into extraction
      f32x8_t acc = gemv_tile(A16, vecH, wave * 16, lane);
      if (lane == 0 || lane == 16) {
        int rb = wave * 16 + (lane ? 8 : 0);
        #pragma unroll
        for (int r = 0; r < 8; ++r) {
          int i = rb + r;
          float rr = (i < M_DIM) ? fmaxf(acc[r] - bs[i], 0.f) : 0.f;
          rH[i] = (_Float16)rr;
        }
      }
    }
    float t_new = 0.5f * (1.f + sqrtf(1.f + 4.f * t * t));   // uniform per thread
    float coef  = (t - 1.f) / t_new;
    t = t_new;
    __syncthreads();
    if (wave < 5) { // g = A^T r ; fused x/y/t update + clip
      f32x8_t acc = gemv_tile(At16, rH, wave * 16, lane);
      if (lane == 0 || lane == 16) {
        int jb = wave * 16 + (lane ? 8 : 0);
        #pragma unroll
        for (int r = 0; r < 8; ++r) {
          int j = jb + r;
          float yv   = ys[j];
          float grad = (yv - xr[j]) + 2.f * acc[r];           // 2*p, p=1
          float xn   = fminf(fmaxf(yv - step * grad, los[j]), ups[j]);
          float yn   = xn + coef * (xn - xs[j]);
          xs[j]   = xn;
          ys[j]   = yn;
          vecH[j] = (_Float16)yn;
        }
      }
    }
    __syncthreads();
  }

  // ---- phase 5: emit x -----------------------------------------------------
  for (int j = tid; j < N_DIM; j += NTHREADS)
    out[(size_t)blk * N_DIM + j] = xs[j];
}

extern "C" void kernel_launch(void* const* d_in, const int* in_sizes, int n_in,
                              void* d_out, int out_size, void* d_ws, size_t ws_size,
                              hipStream_t stream) {
  const float* x_raw = (const float*)d_in[0];
  const float* A     = (const float*)d_in[1];
  const float* b     = (const float*)d_in[2];
  const float* lower = (const float*)d_in[3];
  const float* upper = (const float*)d_in[4];
  float* out = (float*)d_out;
  const int B = in_sizes[0] / N_DIM;   // 8192
  hipLaunchKernelGGL(fista_qp_kernel, dim3(B), dim3(NTHREADS), 0, stream,
                     x_raw, A, b, lower, upper, out);
}
